// LanguageTaskEmbedder_16776142258372
// MI455X (gfx1250) — matrix operand order repaired and checked by reading, not compiled, not run on hardware
//
#include <hip/hip_runtime.h>

// ---------------------------------------------------------------------------
// GRU-style recurrent embedder on MI455X (gfx1250, wave32).
// bf16 operands + f32-accumulate v_wmma_f32_16x16x32_bf16.
// Input GEMM fused into recurrence via K-concat: [h | words_t] @ Wcat.
// Per-step: (256x1536)@(1536x2048) [z,r] then (256x1536)@(1536x1024) [a].
// Weights pre-transposed to N-major bf16 (~13 MiB, resident in 192MB L2).
// GEMM core: 64x128 block tile, 8 wave32s, each wave 32x32 (4 accums),
// K chunked by 64 (8 WMMAs per barrier per wave), double-buffered LDS staged
// with GLOBAL_LOAD_ASYNC_TO_LDS_B128 (ASYNCcnt), hoisted segment pointers.
// ---------------------------------------------------------------------------

typedef __bf16 bf16;
typedef __attribute__((ext_vector_type(16))) __bf16 v16bf;
typedef __attribute__((ext_vector_type(8)))  __bf16 v8bf;
typedef __attribute__((ext_vector_type(8)))  float   v8f;

#define HID   1024
#define TDIM  2048
#define WDIM  512
#define BSZ   256
#define NSEQ  64
#define GATES 8
#define KCAT  (HID + WDIM)   // 1536
#define LDSW  72             // row stride in halves (64 + 8 pad)

// gfx1250 async global->LDS copy (ASYNCcnt). Flip to 0 to fall back.
#define USE_ASYNC_LDS 1

__device__ __forceinline__ float sigmoidf_(float x) {
    return 1.0f / (1.0f + __expf(-x));
}

// 16-byte global -> LDS copy.
__device__ __forceinline__ void ld16_to_lds(bf16* lds, const bf16* g) {
#if USE_ASYNC_LDS
    unsigned l = (unsigned)(uintptr_t)lds;   // low 32 bits = LDS byte address
    asm volatile("global_load_async_to_lds_b128 %0, %1, off"
                 :: "v"(l), "v"(g) : "memory");
#else
    *(v8bf*)lds = *(const v8bf*)g;
#endif
}

__device__ __forceinline__ void lds_copy_wait() {
#if USE_ASYNC_LDS
    asm volatile("s_wait_asynccnt 0x0" ::: "memory");
#endif
}

// ---------------------------------------------------------------------------
// 64(M) x 128(N) bf16 WMMA GEMM tile core, K chunked by 64.
//   A: row-major [M x K], split K: k < K0 from A0 (ld lda0), else A1 (lda1).
//   Bt: row-major [N x K] (B transposed), ld = ldb.
// Block = 256 threads = 8 wave32s. Wave w: M offset 32*(w&1), N offset
// 32*(w>>1); computes 2x2 grid of 16x16 accumulators.
// ---------------------------------------------------------------------------
struct Acc4 { v8f c[2][2]; };

__device__ __forceinline__ void wmma_step(const bf16 (*As)[LDSW],
                                          const bf16 (*Bs)[LDSW],
                                          int ksub, int wmo, int wno, int lane,
                                          Acc4& acc)
{
    const int mlo  = lane & 15;
    const int koff = ksub + (lane >> 4) * 8;   // A frag: K {koff.., 16+koff..}
    const int kb   = ksub + (lane >> 4) * 8;   // B frag: K {kb.., 16+kb..}
    v16bf af[2];
#pragma unroll
    for (int mi = 0; mi < 2; ++mi) {
        const bf16* p = &As[wmo + mi * 16 + mlo][0];
        v8bf lo = *(const v8bf*)(p + koff);
        v8bf hi = *(const v8bf*)(p + 16 + koff);
#pragma unroll
        for (int i = 0; i < 8; ++i) { af[mi][i] = lo[i]; af[mi][8 + i] = hi[i]; }
    }
#pragma unroll
    for (int ni = 0; ni < 2; ++ni) {
        const bf16* p = &Bs[wno + ni * 16 + mlo][0];
        v8bf lo = *(const v8bf*)(p + kb);
        v8bf hi = *(const v8bf*)(p + kb + 16);
        v16bf bf_;
#pragma unroll
        for (int i = 0; i < 8; ++i) { bf_[i] = lo[i]; bf_[8 + i] = hi[i]; }
#pragma unroll
        for (int mi = 0; mi < 2; ++mi)
            acc.c[mi][ni] = __builtin_amdgcn_wmma_f32_16x16x32_bf16(
                false, af[mi], false, bf_, (short)0, acc.c[mi][ni], false, false);
    }
}

__device__ Acc4 gemm_core(const bf16* __restrict__ A0, int lda0, int K0,
                          const bf16* __restrict__ A1, int lda1,
                          const bf16* __restrict__ Bt, int ldb,
                          int rowBase, int colBase, int K)
{
    __shared__ __align__(16) bf16 As[2][64][LDSW];    // ping-pong
    __shared__ __align__(16) bf16 Bs[2][128][LDSW];

    const int tid  = threadIdx.x;
    const int lane = tid & 31;
    const int wave = tid >> 5;
    const int wmo  = (wave & 1) * 32;
    const int wno  = (wave >> 1) * 32;

    const int aRow = tid >> 2, aK = (tid & 3) * 16;  // A: 64x64, 32B/thread
    const int bRow = tid >> 1, bK = (tid & 1) * 32;  // B: 128x64, 64B/thread

    // hoisted per-thread base pointers for both A segments and B
    const bf16* pA0 = A0 + (size_t)(rowBase + aRow) * lda0 + aK;
    const bf16* pA1 = A1 + (size_t)(rowBase + aRow) * lda1 + aK;
    const bf16* pB  = Bt + (size_t)(colBase + bRow) * ldb  + bK;

    Acc4 acc;
#pragma unroll
    for (int mi = 0; mi < 2; ++mi)
#pragma unroll
        for (int ni = 0; ni < 2; ++ni)
            acc.c[mi][ni] = (v8f){0.f,0.f,0.f,0.f,0.f,0.f,0.f,0.f};

    auto stage = [&](int buf, int k0) {
        const bf16* pa = (k0 < K0) ? (pA0 + k0) : (pA1 + (k0 - K0));
        ld16_to_lds(&As[buf][aRow][aK],     pa);
        ld16_to_lds(&As[buf][aRow][aK + 8], pa + 8);
        const bf16* pb = pB + k0;
        ld16_to_lds(&Bs[buf][bRow][bK],      pb);
        ld16_to_lds(&Bs[buf][bRow][bK + 8],  pb + 8);
        ld16_to_lds(&Bs[buf][bRow][bK + 16], pb + 16);
        ld16_to_lds(&Bs[buf][bRow][bK + 24], pb + 24);
    };

    const int nCh = K >> 6;              // 64-wide K chunks
    stage(0, 0);
    for (int i = 0; i < nCh; ++i) {
        lds_copy_wait();                 // my stage(i) landed in LDS
        __syncthreads();                 // everyone's stage(i) + compute(i-1)
        if (i + 1 < nCh) stage((i + 1) & 1, (i + 1) * 64);
        wmma_step(As[i & 1], Bs[i & 1], 0,  wmo, wno, lane, acc);
        wmma_step(As[i & 1], Bs[i & 1], 32, wmo, wno, lane, acc);
    }
    return acc;
}

// Epilogue: C 16x16 f32 layout — VGPR i: lanes0-15 M=i, lanes16-31 M=8+i;
// N = lane%16.
#define EPILOG_BEGIN                                                          \
    const int lane_ = threadIdx.x & 31;                                       \
    const int wave_ = threadIdx.x >> 5;                                       \
    const int wmo_ = (wave_ & 1) * 32;                                        \
    const int wno_ = (wave_ >> 1) * 32;                                       \
    _Pragma("unroll")                                                         \
    for (int mi_ = 0; mi_ < 2; ++mi_)                                         \
    _Pragma("unroll")                                                         \
    for (int ni_ = 0; ni_ < 2; ++ni_) {                                       \
        v8f c_ = acc.c[mi_][ni_];                                             \
        _Pragma("unroll")                                                     \
        for (int i_ = 0; i_ < 8; ++i_) {                                      \
            int row = rowBase + wmo_ + mi_ * 16 + ((lane_ >> 4) << 3) + i_;   \
            int col = colBase + wno_ + ni_ * 16 + (lane_ & 15);               \
            float v = c_[i_];
#define EPILOG_END } }

// ---------------------------------------------------------------------------
// Step kernel 1: ZR = sigmoid([h | w_t] @ Wzr_cat + b_zr)
//   col <  HID : z -> z_f32      col >= HID : r -> rh_bf = bf16(r * h_f32)
// ---------------------------------------------------------------------------
__global__ __launch_bounds__(256) void zr_kernel(
    const bf16* __restrict__ h_bf, const bf16* __restrict__ words_t,
    const bf16* __restrict__ wzr_t, const float* __restrict__ bias,
    const float* __restrict__ h_f32,
    float* __restrict__ z_f32, bf16* __restrict__ rh_bf)
{
    const int rowBase = blockIdx.y * 64;
    const int colBase = blockIdx.x * 128;
    Acc4 acc = gemm_core(h_bf, HID, HID, words_t, WDIM,
                         wzr_t, KCAT, rowBase, colBase, KCAT);
    EPILOG_BEGIN
        float s = sigmoidf_(v + bias[col]);
        if (col < HID) {
            z_f32[(size_t)row * HID + col] = s;
        } else {
            int c2 = col - HID;
            rh_bf[(size_t)row * HID + c2] = (bf16)(s * h_f32[(size_t)row * HID + c2]);
        }
    EPILOG_END
}

// ---------------------------------------------------------------------------
// Step kernel 2: a = tanh([r*h | w_t] @ Wa_cat + b_a);  h = (1-z)*h + z*a
// ---------------------------------------------------------------------------
__global__ __launch_bounds__(256) void upd_kernel(
    const bf16* __restrict__ rh_bf, const bf16* __restrict__ words_t,
    const bf16* __restrict__ wa_t, const float* __restrict__ bias_a,
    const float* __restrict__ z_f32,
    float* __restrict__ h_f32, bf16* __restrict__ h_bf)
{
    const int rowBase = blockIdx.y * 64;
    const int colBase = blockIdx.x * 128;
    Acc4 acc = gemm_core(rh_bf, HID, HID, words_t, WDIM,
                         wa_t, KCAT, rowBase, colBase, KCAT);
    EPILOG_BEGIN
        float a = tanhf(v + bias_a[col]);
        size_t idx = (size_t)row * HID + col;
        float z = z_f32[idx];
        float hn = (1.0f - z) * h_f32[idx] + z * a;
        h_f32[idx] = hn;
        h_bf[idx]  = (bf16)hn;
    EPILOG_END
}

// ---------------------------------------------------------------------------
// Final projection: out = (h @ proj_w + proj_b) * gate[row][col/256]
// ---------------------------------------------------------------------------
__global__ __launch_bounds__(256) void proj_kernel(
    const bf16* __restrict__ h_bf, const bf16* __restrict__ proj_t,
    const float* __restrict__ proj_b, const float* __restrict__ gate_val,
    float* __restrict__ out)
{
    const int rowBase = blockIdx.y * 64;
    const int colBase = blockIdx.x * 128;
    Acc4 acc = gemm_core(h_bf, HID, HID, h_bf, HID,
                         proj_t, HID, rowBase, colBase, HID);
    EPILOG_BEGIN
        out[(size_t)row * TDIM + col] =
            (v + proj_b[col]) * gate_val[row * GATES + (col >> 8)];
    EPILOG_END
}

// ---------------------------------------------------------------------------
// Prep / elementwise kernels
// ---------------------------------------------------------------------------
__global__ void prep_wzr(const float* __restrict__ w_i,
                         const float* __restrict__ w_h,
                         bf16* __restrict__ wzr_t)
{
    int idx = blockIdx.x * 256 + threadIdx.x;
    if (idx >= TDIM * KCAT) return;
    int n = idx / KCAT, k = idx % KCAT;
    float v = (k < HID) ? w_h[(size_t)k * (3 * HID) + n]
                        : w_i[(size_t)(k - HID) * (3 * HID) + n];
    wzr_t[idx] = (bf16)v;
}

__global__ void prep_wa(const float* __restrict__ w_i,
                        const float* __restrict__ w_h,
                        bf16* __restrict__ wa_t)
{
    int idx = blockIdx.x * 256 + threadIdx.x;
    if (idx >= HID * KCAT) return;
    int n = idx / KCAT, k = idx % KCAT;
    float v = (k < HID) ? w_h[(size_t)k * (3 * HID) + (2 * HID + n)]
                        : w_i[(size_t)(k - HID) * (3 * HID) + (2 * HID + n)];
    wa_t[idx] = (bf16)v;
}

__global__ void prep_proj(const float* __restrict__ proj_w,
                          bf16* __restrict__ proj_t)
{
    int idx = blockIdx.x * 256 + threadIdx.x;
    if (idx >= TDIM * HID) return;
    int n = idx / HID, k = idx % HID;
    proj_t[idx] = (bf16)proj_w[(size_t)k * TDIM + n];
}

// words[t][b][d] = (x[b][t] > 0) ? bf16(emb[x[b][t]][d]) : 0
__global__ void embed_kernel(const int* __restrict__ x,
                             const float* __restrict__ emb,
                             bf16* __restrict__ words)
{
    int idx = blockIdx.x * 256 + threadIdx.x;     // group of 4 elements
    if (idx >= (NSEQ * BSZ * WDIM) / 4) return;
    int e  = idx * 4;
    int d  = e & (WDIM - 1);
    int bt = e >> 9;                              // t*256 + b
    int b2 = bt & (BSZ - 1);
    int t  = bt >> 8;
    int tok = x[b2 * NSEQ + t];
    if (tok > 0) {
        const float* src = emb + (size_t)tok * WDIM + d;
#pragma unroll
        for (int i = 0; i < 4; ++i) words[e + i] = (bf16)src[i];
    } else {
#pragma unroll
        for (int i = 0; i < 4; ++i) words[e + i] = (bf16)0.0f;
    }
}

__global__ void init_h(float* __restrict__ h_f32, bf16* __restrict__ h_bf)
{
    int idx = blockIdx.x * 256 + threadIdx.x;
    if (idx >= BSZ * HID) return;
    h_f32[idx] = 0.0f;
    h_bf[idx]  = (bf16)0.0f;
}

// gate[row][g] = round(sigmoid(h_row . gate_w[:,g] + gate_b[g]))
__global__ void gate_kernel(const float* __restrict__ h_f32,
                            const float* __restrict__ gate_w,
                            const float* __restrict__ gate_b,
                            float* __restrict__ gate_val)
{
    int idx = blockIdx.x * 256 + threadIdx.x;
    if (idx >= BSZ * GATES) return;
    int row = idx >> 3, g = idx & 7;
    float s = 0.0f;
    for (int k = 0; k < HID; ++k)
        s += h_f32[(size_t)row * HID + k] * gate_w[(size_t)k * GATES + g];
    gate_val[idx] = rintf(sigmoidf_(s + gate_b[g]));
}

// ---------------------------------------------------------------------------
extern "C" void kernel_launch(void* const* d_in, const int* in_sizes, int n_in,
                              void* d_out, int out_size, void* d_ws, size_t ws_size,
                              hipStream_t stream)
{
    const int*   x      = (const int*)  d_in[0];
    const float* emb    = (const float*)d_in[1];
    const float* w_i    = (const float*)d_in[2];
    const float* w_h    = (const float*)d_in[3];
    const float* b      = (const float*)d_in[4];
    const float* proj_w = (const float*)d_in[5];
    const float* proj_b = (const float*)d_in[6];
    const float* gate_w = (const float*)d_in[7];
    const float* gate_b = (const float*)d_in[8];
    float* out = (float*)d_out;

    // workspace carve-up (~32 MiB total)
    size_t off = 0;
    auto carve = [&](size_t bytes) {
        void* p = (char*)d_ws + off;
        off += (bytes + 255) & ~(size_t)255;
        return p;
    };
    bf16*  words   = (bf16*) carve((size_t)NSEQ * BSZ * WDIM * 2); // 16 MiB
    bf16*  wzr_t   = (bf16*) carve((size_t)TDIM * KCAT * 2);       //  6 MiB
    bf16*  wa_t    = (bf16*) carve((size_t)HID  * KCAT * 2);       //  3 MiB
    bf16*  proj_t  = (bf16*) carve((size_t)TDIM * HID  * 2);       //  4 MiB
    float* h_f32   = (float*)carve((size_t)BSZ * HID * 4);         //  1 MiB
    bf16*  h_bf    = (bf16*) carve((size_t)BSZ * HID * 2);
    float* z_f32   = (float*)carve((size_t)BSZ * HID * 4);
    bf16*  rh_bf   = (bf16*) carve((size_t)BSZ * HID * 2);
    float* gate_v  = (float*)carve((size_t)BSZ * GATES * 4);

    // ---- prep ----
    prep_wzr   <<<(TDIM * KCAT + 255) / 256, 256, 0, stream>>>(w_i, w_h, wzr_t);
    prep_wa    <<<(HID  * KCAT + 255) / 256, 256, 0, stream>>>(w_i, w_h, wa_t);
    prep_proj  <<<(TDIM * HID  + 255) / 256, 256, 0, stream>>>(proj_w, proj_t);
    embed_kernel<<<(NSEQ * BSZ * WDIM / 4 + 255) / 256, 256, 0, stream>>>(x, emb, words);
    init_h     <<<(BSZ * HID + 255) / 256, 256, 0, stream>>>(h_f32, h_bf);

    // ---- recurrence: 64 steps, 2 dependent GEMM kernels per step ----
    for (int t = 0; t < NSEQ; ++t) {
        const bf16* wt = words + (size_t)t * BSZ * WDIM;
        zr_kernel <<<dim3(TDIM / 128, BSZ / 64), 256, 0, stream>>>(
            h_bf, wt, wzr_t, b, h_f32, z_f32, rh_bf);
        upd_kernel<<<dim3(HID / 128, BSZ / 64), 256, 0, stream>>>(
            rh_bf, wt, wa_t, b + 2 * HID, z_f32, h_f32, h_bf);
    }

    // ---- gates + fused projection ----
    gate_kernel<<<(BSZ * GATES + 255) / 256, 256, 0, stream>>>(
        h_f32, gate_w, gate_b, gate_v);
    proj_kernel<<<dim3(TDIM / 128, BSZ / 64), 256, 0, stream>>>(
        h_bf, proj_t, proj_b, gate_v, out);
}
